// AttentionSub_43361989820791
// MI455X (gfx1250) — matrix operand order repaired and checked by reading, not compiled, not run on hardware
//
#include <hip/hip_runtime.h>
#include <hip/hip_bf16.h>
#include <stdint.h>

typedef __bf16 bf16;
typedef __attribute__((ext_vector_type(16))) __bf16 v16bf;
typedef __attribute__((ext_vector_type(8)))  __bf16 v8bf;
typedef __attribute__((ext_vector_type(8)))  float  v8f;

#define B_    64
#define N_    197
#define C_    768
#define H_    12
#define HD_   64
#define NROW  (B_ * N_)      /* 12608 */
#define NQP   208            /* 13 * 16: padded key/query rows */
#define NPAD  224            /* 7 * 32: padded key columns in score tile */
#define MT    (NROW / 16)    /* 788 row tiles */

static __device__ inline v8f zero8() {
  v8f z;
#pragma unroll
  for (int i = 0; i < 8; ++i) z[i] = 0.0f;
  return z;
}

// A-matrix (16x32 bf16) per-lane pack: elems 0-7 -> K = 16*(lane/16)+j,
// elems 8-15 -> K = 16*(lane/16)+8+j  (contiguous k order, matching the B loads).
static __device__ inline v16bf pack_a_f32(const float* p0, const float* p1) {
  v16bf a;
#pragma unroll
  for (int i = 0; i < 8; ++i) { a[i] = (bf16)p0[i]; a[i + 8] = (bf16)p1[i]; }
  return a;
}

static __device__ inline v16bf pack_a_bf16(const bf16* p0, const bf16* p1) {
  union { v16bf v; v8bf h[2]; } u;
  u.h[0] = *(const v8bf*)p0;
  u.h[1] = *(const v8bf*)p1;
  return u.v;
}

static __device__ inline v8f wmma_bf16(v16bf a, v16bf b, v8f c) {
  // 8 args: (neg_a, A, neg_b, B, c_mod, C, reuse_a, reuse_b)
  return __builtin_amdgcn_wmma_f32_16x16x32_bf16(false, a, false, b, (short)0, c,
                                                 false, false);
}

// Relative-position table index (mirrors rel_pos_indices: s=14, max_rel=14).
static __device__ inline void relidx(int qg, int kg, int& iv, int& ih) {
  if (qg > 0 && qg < N_ && kg > 0 && kg < N_) {
    const int rq = qg - 1, rk = kg - 1;
    int dv = rk / 14 - rq / 14;
    int dh = rk % 14 - rq % 14;
    dv = dv < -14 ? -14 : (dv > 14 ? 14 : dv);
    dh = dh < -14 ? -14 : (dh > 14 ? 14 : dh);
    iv = dv + 15;
    ih = dh + 15;
  } else {
    iv = 0;
    ih = 0;
  }
}

// ---------------------------------------------------------------- prep kernels
__global__ void k_zero_u32(uint32_t* p, int n) {
  const int i = blockIdx.x * blockDim.x + threadIdx.x;
  if (i < n) p[i] = 0u;
}

__global__ void k_transpose_bf16(const float* __restrict__ src, bf16* __restrict__ dst,
                                 int rows, int cols) {
  const int i = blockIdx.x * blockDim.x + threadIdx.x;
  if (i >= rows * cols) return;
  const int r = i / cols, c = i % cols;
  dst[c * rows + r] = (bf16)src[i];
}

__global__ void k_tabs(const float* __restrict__ tvk, const float* __restrict__ thk,
                       const float* __restrict__ tvv, const float* __restrict__ thv,
                       bf16* __restrict__ tkv, bf16* __restrict__ tkh,
                       bf16* __restrict__ tvvT, bf16* __restrict__ thvT) {
  const int i = blockIdx.x * blockDim.x + threadIdx.x;
  if (i >= 32 * 64) return;
  const int r = i >> 6, d = i & 63;
  float a = 0.0f, b = 0.0f, c = 0.0f, e = 0.0f;
  if (r < 30) {
    a = tvk[r * 64 + d];
    b = thk[r * 64 + d];
    c = tvv[r * 64 + d];
    e = thv[r * 64 + d];
  }
  tkv[r * 64 + d] = (bf16)a;   // row-major, padded to 32 rows
  tkh[r * 64 + d] = (bf16)b;
  tvvT[d * 32 + r] = (bf16)c;  // transposed [64][32] for B-operand loads
  thvT[d * 32 + r] = (bf16)e;
}

// ---------------------------------------------------------------- QKV GEMM
// [12608 x 768] @ [768 x 2304], one wave per 16x64 output slab.
__global__ __launch_bounds__(32) void k_qkv_gemm(
    const float* __restrict__ x, const bf16* __restrict__ wT,
    bf16* __restrict__ qws, bf16* __restrict__ kws, bf16* __restrict__ vtws) {
  const int tm = blockIdx.x;   // 0..787
  const int tc4 = blockIdx.y;  // 0..35 (64 columns each)
  const int lane = threadIdx.x;
  const int l15 = lane & 15;
  const int hf = lane >> 4;

  v8f acc[4];
#pragma unroll
  for (int t = 0; t < 4; ++t) acc[t] = zero8();

  const float* xrow = x + (size_t)(tm * 16 + l15) * C_;
  for (int kk = 0; kk < C_ / 32; ++kk) {
    const int d0 = kk * 32 + hf * 16;
    const v16bf a = pack_a_f32(xrow + d0, xrow + d0 + 8);
#pragma unroll
    for (int t = 0; t < 4; ++t) {
      const v16bf b = *(const v16bf*)(wT + (size_t)(tc4 * 64 + t * 16 + l15) * C_ +
                                      kk * 32 + hf * 16);
      acc[t] = wmma_bf16(a, b, acc[t]);
    }
  }

#pragma unroll
  for (int t = 0; t < 4; ++t) {
    const int c = tc4 * 64 + t * 16 + l15;
    const int which = c / C_;           // 0=q 1=k 2=v
    const int hh = (c % C_) / HD_;
    const int d = c % HD_;
#pragma unroll
    for (int r = 0; r < 8; ++r) {
      const int row = tm * 16 + r + 8 * hf;  // < 12608 always
      const int bb = row / N_, nn = row % N_;
      const bf16 val = (bf16)acc[t][r];
      const size_t bh = (size_t)bb * H_ + hh;
      if (which == 0)
        qws[(bh * NQP + nn) * HD_ + d] = val;
      else if (which == 1)
        kws[(bh * NQP + nn) * HD_ + d] = val;
      else
        vtws[(bh * HD_ + d) * NPAD + nn] = val;  // V stored transposed
    }
  }
}

// ---------------------------------------------------------------- fused attention
// One wave per (batch*head, 16-query tile). Whole score row lives in LDS.
__global__ __launch_bounds__(32) void k_attn(
    const bf16* __restrict__ qws, const bf16* __restrict__ kws,
    const bf16* __restrict__ vtws, const bf16* __restrict__ tkv,
    const bf16* __restrict__ tkh, const bf16* __restrict__ tvvT,
    const bf16* __restrict__ thvT, bf16* __restrict__ aout) {
  __shared__ float S[16][NPAD];  // score / prob tile
  __shared__ float Pv[16][32];   // q . tab_vk[r]
  __shared__ float Ph[16][32];   // q . tab_hk[r]
  __shared__ float Wv[16][32];   // sum of probs per idx_v bin
  __shared__ float Wh[16][32];   // sum of probs per idx_h bin

  const int qt = blockIdx.x;   // 0..12
  const int bh = blockIdx.y;   // 0..767
  const int bidx = bh / H_;
  const int hidx = bh % H_;
  const int lane = threadIdx.x;
  const int l15 = lane & 15;
  const int hf = lane >> 4;
  const int q0 = qt * 16;

  const bf16* Q = qws + (size_t)bh * NQP * HD_;
  const bf16* K = kws + (size_t)bh * NQP * HD_;
  const bf16* VT = vtws + (size_t)bh * HD_ * NPAD;

  for (int i = lane; i < 16 * 32; i += 32) {
    (&Wv[0][0])[i] = 0.0f;
    (&Wh[0][0])[i] = 0.0f;
  }

  // Q tile as WMMA A operand (K-dim = hd = 64 -> two K=32 slices).
  const bf16* qrow = Q + (size_t)(q0 + l15) * HD_;
  const v16bf aq0 = pack_a_bf16(qrow + hf * 16, qrow + hf * 16 + 8);
  const v16bf aq1 = pack_a_bf16(qrow + 32 + hf * 16, qrow + 32 + hf * 16 + 8);

  // P_v = Q @ tab_vk^T, P_h = Q @ tab_hk^T (30 rows padded to 32).
#pragma unroll
  for (int nt = 0; nt < 2; ++nt) {
    v8f av = zero8(), ah = zero8();
    const bf16* tr = tkv + (size_t)(nt * 16 + l15) * HD_;
    av = wmma_bf16(aq0, *(const v16bf*)(tr + hf * 16), av);
    av = wmma_bf16(aq1, *(const v16bf*)(tr + 32 + hf * 16), av);
    const bf16* th = tkh + (size_t)(nt * 16 + l15) * HD_;
    ah = wmma_bf16(aq0, *(const v16bf*)(th + hf * 16), ah);
    ah = wmma_bf16(aq1, *(const v16bf*)(th + 32 + hf * 16), ah);
#pragma unroll
    for (int r = 0; r < 8; ++r) {
      Pv[r + 8 * hf][nt * 16 + l15] = av[r];
      Ph[r + 8 * hf][nt * 16 + l15] = ah[r];
    }
  }
  __syncthreads();

  // Scores: S = (Q K^T + gather(Pv,Ph)) * scale over 13 key tiles.
  for (int kt = 0; kt < NQP / 16; ++kt) {
    const int k0 = kt * 16;
    v8f acc = zero8();
    const bf16* krow = K + (size_t)(k0 + l15) * HD_;
    acc = wmma_bf16(aq0, *(const v16bf*)(krow + hf * 16), acc);
    acc = wmma_bf16(aq1, *(const v16bf*)(krow + 32 + hf * 16), acc);
    const int kg = k0 + l15;
#pragma unroll
    for (int r = 0; r < 8; ++r) {
      const int m = r + 8 * hf;
      int iv, ih;
      relidx(q0 + m, kg, iv, ih);
      const float bias = Pv[m][iv] + Ph[m][ih];
      S[m][kg] = (acc[r] + bias) * 0.125f;  // scale = hd^-0.5
    }
  }
  __syncthreads();

  // Softmax, one query row per lane (lanes 16..31 idle; reconverges below).
  if (lane < 16) {
    float mx = -1e30f;
    for (int c = 0; c < N_; ++c) mx = fmaxf(mx, S[lane][c]);
    float sum = 0.0f;
    for (int c = 0; c < N_; ++c) {
      const float e = __expf(S[lane][c] - mx);
      S[lane][c] = e;
      sum += e;
    }
    const float inv = 1.0f / sum;
    for (int c = 0; c < N_; ++c) S[lane][c] *= inv;
    for (int c = N_; c < NPAD; ++c) S[lane][c] = 0.0f;
  }
  __syncthreads();

  // Scatter probs into 30 rel-pos bins per query (LDS float atomics).
  {
    const int m = l15;
    const int qg = q0 + m;
    if (qg < N_) {
      const int cb = hf ? 100 : 0;
      const int ce = hf ? N_ : 100;
      for (int c = cb; c < ce; ++c) {
        const float p = S[m][c];
        int iv, ih;
        relidx(qg, c, iv, ih);
        atomicAdd(&Wv[m][iv], p);
        atomicAdd(&Wh[m][ih], p);
      }
    }
  }
  __syncthreads();

  // O = P @ V + Wv @ tab_vv + Wh @ tab_hv, chained into one accumulator.
#pragma unroll
  for (int dt = 0; dt < 4; ++dt) {
    v8f acc = zero8();
#pragma unroll
    for (int kc = 0; kc < NPAD / 32; ++kc) {
      const v16bf a =
          pack_a_f32(&S[l15][kc * 32 + hf * 16], &S[l15][kc * 32 + hf * 16 + 8]);
      const v16bf b =
          *(const v16bf*)(VT + (size_t)(dt * 16 + l15) * NPAD + kc * 32 + hf * 16);
      acc = wmma_bf16(a, b, acc);
    }
    {
      v16bf a = pack_a_f32(&Wv[l15][hf * 16], &Wv[l15][hf * 16 + 8]);
      v16bf b = *(const v16bf*)(tvvT + (size_t)(dt * 16 + l15) * 32 + hf * 16);
      acc = wmma_bf16(a, b, acc);
      a = pack_a_f32(&Wh[l15][hf * 16], &Wh[l15][hf * 16 + 8]);
      b = *(const v16bf*)(thvT + (size_t)(dt * 16 + l15) * 32 + hf * 16);
      acc = wmma_bf16(a, b, acc);
    }
#pragma unroll
    for (int r = 0; r < 8; ++r) {
      const int m = r + 8 * hf;
      const int qg = q0 + m;
      if (qg < N_) {
        const int d = dt * 16 + l15;
        aout[((size_t)bidx * N_ + qg) * C_ + hidx * HD_ + d] = (bf16)acc[r];
      }
    }
  }
}

// ---------------------------------------------------------------- projection GEMM
__global__ __launch_bounds__(32) void k_proj_gemm(
    const bf16* __restrict__ aout, const bf16* __restrict__ wT,
    const float* __restrict__ bias, float* __restrict__ out) {
  const int tm = blockIdx.x;   // 0..787
  const int tc4 = blockIdx.y;  // 0..11
  const int lane = threadIdx.x;
  const int l15 = lane & 15;
  const int hf = lane >> 4;

  v8f acc[4];
#pragma unroll
  for (int t = 0; t < 4; ++t) acc[t] = zero8();

  const bf16* arow = aout + (size_t)(tm * 16 + l15) * C_;
  for (int kk = 0; kk < C_ / 32; ++kk) {
    const int d0 = kk * 32 + hf * 16;
    const v16bf a = pack_a_bf16(arow + d0, arow + d0 + 8);
#pragma unroll
    for (int t = 0; t < 4; ++t) {
      const v16bf b = *(const v16bf*)(wT + (size_t)(tc4 * 64 + t * 16 + l15) * C_ +
                                      kk * 32 + hf * 16);
      acc[t] = wmma_bf16(a, b, acc[t]);
    }
  }

#pragma unroll
  for (int t = 0; t < 4; ++t) {
    const int c = tc4 * 64 + t * 16 + l15;
    const float bc = bias[c];
#pragma unroll
    for (int r = 0; r < 8; ++r) {
      const int row = tm * 16 + r + 8 * hf;
      out[(size_t)row * C_ + c] = acc[t][r] + bc;
    }
  }
}

// ---------------------------------------------------------------- launcher
extern "C" void kernel_launch(void* const* d_in, const int* in_sizes, int n_in,
                              void* d_out, int out_size, void* d_ws, size_t ws_size,
                              hipStream_t stream) {
  (void)in_sizes; (void)n_in; (void)out_size; (void)ws_size;
  const float* x = (const float*)d_in[0];
  const float* w_qkv = (const float*)d_in[1];
  const float* tab_vk = (const float*)d_in[2];
  const float* tab_hk = (const float*)d_in[3];
  const float* tab_vv = (const float*)d_in[4];
  const float* tab_hv = (const float*)d_in[5];
  const float* w_proj = (const float*)d_in[6];
  const float* b_proj = (const float*)d_in[7];
  float* out = (float*)d_out;

  char* ws = (char*)d_ws;
  size_t off = 0;
  auto take = [&](size_t elems) {
    char* p = ws + off;
    off += (elems * sizeof(bf16) + 255) & ~(size_t)255;
    return (bf16*)p;
  };
  bf16* wqkvT = take((size_t)3 * C_ * C_);        // [2304][768]
  bf16* wprojT = take((size_t)C_ * C_);           // [768][768]
  bf16* tkv = take(32 * 64);
  bf16* tkh = take(32 * 64);
  bf16* tvvT = take(64 * 32);
  bf16* thvT = take(64 * 32);
  bf16* qws = take((size_t)B_ * H_ * NQP * HD_);
  bf16* kws = take((size_t)B_ * H_ * NQP * HD_);
  bf16* vtws = take((size_t)B_ * H_ * HD_ * NPAD);
  bf16* aout = take((size_t)NROW * C_);

  {
    const int n1 = (B_ * H_ * NQP * HD_) / 2;
    k_zero_u32<<<(n1 + 255) / 256, 256, 0, stream>>>((uint32_t*)qws, n1);
    k_zero_u32<<<(n1 + 255) / 256, 256, 0, stream>>>((uint32_t*)kws, n1);
    const int n2 = (B_ * H_ * HD_ * NPAD) / 2;
    k_zero_u32<<<(n2 + 255) / 256, 256, 0, stream>>>((uint32_t*)vtws, n2);
  }
  k_transpose_bf16<<<(C_ * 3 * C_ + 255) / 256, 256, 0, stream>>>(w_qkv, wqkvT, C_,
                                                                  3 * C_);
  k_transpose_bf16<<<(C_ * C_ + 255) / 256, 256, 0, stream>>>(w_proj, wprojT, C_, C_);
  k_tabs<<<8, 256, 0, stream>>>(tab_vk, tab_hk, tab_vv, tab_hv, tkv, tkh, tvvT, thvT);

  k_qkv_gemm<<<dim3(MT, (3 * C_) / 64), 32, 0, stream>>>(x, wqkvT, qws, kws, vtws);
  k_attn<<<dim3(NQP / 16, B_ * H_), 32, 0, stream>>>(qws, kws, vtws, tkv, tkh, tvvT,
                                                     thvT, aout);
  k_proj_gemm<<<dim3(MT, C_ / 64), 32, 0, stream>>>(aout, wprojT, b_proj, out);
}
